// GravityNet_39591008535097
// MI455X (gfx1250) — matrix operand order repaired and checked by reading, not compiled, not run on hardware
//
#include <hip/hip_runtime.h>
#include <hip/hip_bf16.h>

typedef __attribute__((ext_vector_type(16))) _Float16 v16h;
typedef __attribute__((ext_vector_type(8)))  float    v8f;
typedef __attribute__((ext_vector_type(4)))  unsigned u32x4;
typedef __attribute__((ext_vector_type(2)))  unsigned u32x2;
typedef __attribute__((ext_vector_type(4)))  float    f32x4;

#define MAXR     224           // max rows per segment handled in LDS (true max ~170)
#define XSTRIDE  104           // halves per X row (96 data + pad, bank spread)
#define HSTRIDE  520           // halves per H row (512 data + pad, bank spread)
#define ZSTRIDE  72            // halves per Z-group row (64 data + pad)
#define EPSV     1e-5f

// ---- dynamic LDS layout (bytes) ----
#define OFF_CST   0                       // 162 floats (Ws 128, bs 32, -bm*om[2])
#define OFF_SUM1  768
#define OFF_SQ1   (768 + 2048)
#define OFF_SUM2  (768 + 4096)
#define OFF_SQ2   (768 + 4096 + 256)
#define OFF_X     5376                    // 224*104*2 = 46592 B (reused as Z buffer)
#define OFF_H     51968                   // 224*520*2 = 232960 B
#define LDS_BYTES 284928                  // 278.25 KB <= 320 KB WGP LDS

union Frag  { v16h v; u32x4 q[2]; };
union H8    { u32x4 q; _Float16 h[8]; };
union H4    { u32x2 u; _Float16 h[4]; };
union Pack2 { _Float16 h[2]; unsigned u; };

__device__ __forceinline__ unsigned pack2(float a, float b) {
    Pack2 p; p.h[0] = (_Float16)a; p.h[1] = (_Float16)b; return p.u;
}

// Convert W1 (512x96) and W2 (1024x512) to f16 copies in workspace.
__global__ __launch_bounds__(256) void prep_weights(const float* __restrict__ W1,
                                                    const float* __restrict__ W2,
                                                    _Float16* __restrict__ W1h,
                                                    _Float16* __restrict__ W2h) {
    int i = blockIdx.x * 256 + threadIdx.x;
    if (i < 512 * 96) W1h[i] = (_Float16)W1[i];
    int j = i - 512 * 96;
    if (j >= 0 && j < 1024 * 512) W2h[j] = (_Float16)W2[j];
}

// One workgroup (8 wave32) per segment: fused emb -> GEMM1 -> segBN -> relu
// -> GEMM2 (in 64-channel groups) -> segBN -> relu -> out.
__global__ __launch_bounds__(256, 1)
void gravity_fused(const float* __restrict__ h_state,
                   const int*   __restrict__ sse,
                   const float* __restrict__ cbr,
                   const float* __restrict__ bmass,
                   const float* __restrict__ omass,
                   const float* __restrict__ Wsm,
                   const float* __restrict__ bsv,
                   const float* __restrict__ g1,  const float* __restrict__ be1,
                   const float* __restrict__ g2,  const float* __restrict__ be2,
                   const _Float16* __restrict__ W1h,
                   const _Float16* __restrict__ W2h,
                   float* __restrict__ out) {
    extern __shared__ __align__(16) char smem[];
    float*    cst  = (float*)(smem + OFF_CST);
    float*    sum1 = (float*)(smem + OFF_SUM1);
    float*    sq1  = (float*)(smem + OFF_SQ1);
    float*    sum2 = (float*)(smem + OFF_SUM2);
    float*    sq2  = (float*)(smem + OFF_SQ2);
    _Float16* Xh   = (_Float16*)(smem + OFF_X);
    _Float16* Hh   = (_Float16*)(smem + OFF_H);
    _Float16* Zh   = Xh;                       // X region reused for z2 groups
    unsigned* Xu   = (unsigned*)Xh;

    const int tid   = threadIdx.x;
    const int seg   = blockIdx.x;
    const int start = sse[2 * seg];
    const int end   = sse[2 * seg + 1];
    int cnt = end - start; if (cnt > MAXR) cnt = MAXR;
    const int   RT     = (cnt + 15) >> 4;
    const int   padR   = RT << 4;
    const float invCnt = 1.0f / (float)cnt;

    // ---- stage constants, zero layer-1 stats ----
    if (tid < 128)       cst[tid] = Wsm[tid];
    else if (tid < 160)  cst[tid] = bsv[tid - 128];
    else if (tid == 160) { float b = bmass[0]; cst[160] = -b * omass[0]; cst[161] = -b * omass[1]; }
    if (tid < 512) { sum1[tid] = 0.f; sq1[tid] = 0.f; }
    __syncthreads();

    // ---- Stage A: build X = [emb | h_state] as f16 rows in LDS (zero pad rows) ----
    for (int r = tid; r < padR; r += 256) {
        unsigned* row = Xu + r * (XSTRIDE / 2);
        if (r < cnt) {
            const int g = start + r;
            const float* cb = cbr + (size_t)g * 4;
            float c00 = cb[0], c01 = cb[1], c10 = cb[2], c11 = cb[3];
            float d20 = c00 * c00 + c10 * c10;
            float d21 = c01 * c01 + c11 * c11;
            float r0 = cst[160] * c00 / d20;
            float r1 = cst[160] * c10 / d20;
            float r2 = cst[161] * c01 / d21;
            float r3 = cst[161] * c11 / d21;
            float e[32];
            #pragma unroll
            for (int j = 0; j < 32; ++j)
                e[j] = cst[128 + j] + cst[4*j]*r0 + cst[4*j+1]*r1 + cst[4*j+2]*r2 + cst[4*j+3]*r3;
            #pragma unroll
            for (int j = 0; j < 16; ++j) row[j] = pack2(e[2*j], e[2*j+1]);
            const float* hs = h_state + (size_t)g * 64;
            #pragma unroll
            for (int j = 0; j < 32; ++j) row[16 + j] = pack2(hs[2*j], hs[2*j+1]);
        } else {
            #pragma unroll
            for (int j = 0; j < 48; ++j) row[j] = 0u;
        }
    }
    __syncthreads();

    const int  lane  = tid & 31;
    const int  wave  = tid >> 5;
    const int  lh    = lane & 15;
    const bool lo    = lane < 16;

    // ---- GEMM1: z1 tiles via v_wmma_f32_16x16x32_f16.
    // Each wave handles a PAIR of N-tiles sharing one A fragment: halves LDS
    // traffic and interleaves independent WMMAs to hide the RAW-hazard window.
    {
        const int nt = RT * 16;                 // pairs of column tiles
        for (int t = wave; t < nt; t += 8) {
            const int tm  = t >> 4;
            const int tn0 = (t & 15) << 1;      // tn0, tn0+1
            v8f acc0 = {0.f,0.f,0.f,0.f,0.f,0.f,0.f,0.f};
            v8f acc1 = {0.f,0.f,0.f,0.f,0.f,0.f,0.f,0.f};
            const _Float16* ap  = Xh + (tm * 16 + lh) * XSTRIDE + (lo ? 0 : 8);
            const _Float16* bp0 = W1h + (tn0 * 16 + lh) * 96 + (lo ? 0 : 8);
            const _Float16* bp1 = bp0 + 16 * 96;
            #pragma unroll
            for (int kb = 0; kb < 3; ++kb) {
                Frag a, b0, b1;
                a.q[0]  = *(const u32x4*)(ap  + kb * 32);
                a.q[1]  = *(const u32x4*)(ap  + kb * 32 + 16);
                b0.q[0] = *(const u32x4*)(bp0 + kb * 32);
                b0.q[1] = *(const u32x4*)(bp0 + kb * 32 + 16);
                b1.q[0] = *(const u32x4*)(bp1 + kb * 32);
                b1.q[1] = *(const u32x4*)(bp1 + kb * 32 + 16);
                acc0 = __builtin_amdgcn_wmma_f32_16x16x32_f16(false, a.v, false, b0.v,
                                                              (short)0, acc0, false, false);
                acc1 = __builtin_amdgcn_wmma_f32_16x16x32_f16(false, a.v, false, b1.v,
                                                              (short)0, acc1, false, false);
            }
            const int mb = tm * 16 + (lo ? 0 : 8);
            {
                const int col = tn0 * 16 + lh;
                float s = 0.f, s2 = 0.f;
                #pragma unroll
                for (int j = 0; j < 8; ++j) {
                    float v = acc0[j];
                    s += v; s2 += v * v;
                    Hh[(mb + j) * HSTRIDE + col] = (_Float16)v;
                }
                atomicAdd(&sum1[col], s);
                atomicAdd(&sq1[col], s2);
            }
            {
                const int col = (tn0 + 1) * 16 + lh;
                float s = 0.f, s2 = 0.f;
                #pragma unroll
                for (int j = 0; j < 8; ++j) {
                    float v = acc1[j];
                    s += v; s2 += v * v;
                    Hh[(mb + j) * HSTRIDE + col] = (_Float16)v;
                }
                atomicAdd(&sum1[col], s);
                atomicAdd(&sq1[col], s2);
            }
        }
    }
    __syncthreads();

    // ---- layer-1 BN params: sum1 <- scale, sq1 <- shift (b1 cancels in z-mean) ----
    if (tid < 512) {
        float mean = sum1[tid] * invCnt;
        float var  = sq1[tid] * invCnt - mean * mean;
        float sc   = rsqrtf(var + EPSV) * g1[tid];
        sum1[tid]  = sc;
        sq1[tid]   = be1[tid] - mean * sc;
    }
    __syncthreads();

    // ---- normalize + relu H in place; zero pad rows so they vanish in GEMM2 ----
    for (int idx = tid; idx < padR * 64; idx += 256) {
        const int r  = idx >> 6;
        const int cb = (idx & 63) << 3;
        u32x4* p = (u32x4*)(Hh + r * HSTRIDE + cb);
        if (r < cnt) {
            H8 u; u.q = *p;
            #pragma unroll
            for (int e = 0; e < 8; ++e) {
                float v = (float)u.h[e] * sum1[cb + e] + sq1[cb + e];
                u.h[e] = (_Float16)(v > 0.f ? v : 0.f);
            }
            *p = u.q;
        } else {
            u32x4 z = {};
            *p = z;
        }
    }
    __syncthreads();

    // ---- GEMM2 + BN + relu, streamed in 16 groups of 64 output channels ----
    for (int gb = 0; gb < 16; ++gb) {
        if (tid < 64) { sum2[tid] = 0.f; sq2[tid] = 0.f; }
        __syncthreads();

        // pairs: (tm, q0={0,2}); each wave computes two column tiles per A frag
        const int nt = RT * 2;
        for (int t = wave; t < nt; t += 8) {
            const int tm = t >> 1;
            const int q0 = (t & 1) << 1;        // q0, q0+1
            v8f acc0 = {0.f,0.f,0.f,0.f,0.f,0.f,0.f,0.f};
            v8f acc1 = {0.f,0.f,0.f,0.f,0.f,0.f,0.f,0.f};
            const _Float16* ap  = Hh + (tm * 16 + lh) * HSTRIDE + (lo ? 0 : 8);
            const _Float16* bp0 = W2h + (size_t)(gb * 64 + q0 * 16 + lh) * 512 + (lo ? 0 : 8);
            const _Float16* bp1 = bp0 + 16 * 512;
            #pragma unroll 4
            for (int kb = 0; kb < 16; ++kb) {
                Frag a, b0, b1;
                a.q[0]  = *(const u32x4*)(ap  + kb * 32);
                a.q[1]  = *(const u32x4*)(ap  + kb * 32 + 16);
                b0.q[0] = *(const u32x4*)(bp0 + kb * 32);
                b0.q[1] = *(const u32x4*)(bp0 + kb * 32 + 16);
                b1.q[0] = *(const u32x4*)(bp1 + kb * 32);
                b1.q[1] = *(const u32x4*)(bp1 + kb * 32 + 16);
                acc0 = __builtin_amdgcn_wmma_f32_16x16x32_f16(false, a.v, false, b0.v,
                                                              (short)0, acc0, false, false);
                acc1 = __builtin_amdgcn_wmma_f32_16x16x32_f16(false, a.v, false, b1.v,
                                                              (short)0, acc1, false, false);
            }
            const int mb = tm * 16 + (lo ? 0 : 8);
            {
                const int colg = q0 * 16 + lh;
                float s = 0.f, s2 = 0.f;
                #pragma unroll
                for (int j = 0; j < 8; ++j) {
                    float v = acc0[j];
                    s += v; s2 += v * v;
                    Zh[(mb + j) * ZSTRIDE + colg] = (_Float16)v;
                }
                atomicAdd(&sum2[colg], s);
                atomicAdd(&sq2[colg], s2);
            }
            {
                const int colg = (q0 + 1) * 16 + lh;
                float s = 0.f, s2 = 0.f;
                #pragma unroll
                for (int j = 0; j < 8; ++j) {
                    float v = acc1[j];
                    s += v; s2 += v * v;
                    Zh[(mb + j) * ZSTRIDE + colg] = (_Float16)v;
                }
                atomicAdd(&sum2[colg], s);
                atomicAdd(&sq2[colg], s2);
            }
        }
        __syncthreads();

        if (tid < 64) {
            const int c = gb * 64 + tid;
            float mean = sum2[tid] * invCnt;
            float var  = sq2[tid] * invCnt - mean * mean;
            float sc   = rsqrtf(var + EPSV) * g2[c];
            sum2[tid]  = sc;
            sq2[tid]   = be2[c] - mean * sc;
        }
        __syncthreads();

        for (int idx = tid; idx < cnt * 16; idx += 256) {
            const int r  = idx >> 4;
            const int c4 = (idx & 15) << 2;
            H4 zz; zz.u = *(const u32x2*)(Zh + r * ZSTRIDE + c4);
            f32x4 o;
            #pragma unroll
            for (int e = 0; e < 4; ++e) {
                float v = (float)zz.h[e] * sum2[c4 + e] + sq2[c4 + e];
                o[e] = v > 0.f ? v : 0.f;
            }
            *(f32x4*)(out + (size_t)(start + r) * 1024 + gb * 64 + c4) = o;
        }
        __syncthreads();   // protect Zh / sum2 before next group
    }
}

extern "C" void kernel_launch(void* const* d_in, const int* in_sizes, int n_in,
                              void* d_out, int out_size, void* d_ws, size_t ws_size,
                              hipStream_t stream) {
    const float* h_state = (const float*)d_in[0];
    const int*   sse     = (const int*)  d_in[1];
    const float* cbr     = (const float*)d_in[2];
    const float* bm      = (const float*)d_in[3];
    const float* om      = (const float*)d_in[4];
    const float* Wsm     = (const float*)d_in[5];
    const float* bsv     = (const float*)d_in[6];
    const float* W1      = (const float*)d_in[7];
    // d_in[8] = b1, d_in[12] = b2: row-constant shifts cancel exactly in segment BN
    const float* g1      = (const float*)d_in[9];
    const float* be1     = (const float*)d_in[10];
    const float* W2      = (const float*)d_in[11];
    const float* g2      = (const float*)d_in[13];
    const float* be2     = (const float*)d_in[14];
    float* out = (float*)d_out;

    _Float16* W1h = (_Float16*)d_ws;            // 512*96  f16
    _Float16* W2h = W1h + 512 * 96;             // 1024*512 f16 (total ~1.15 MB)

    const int nconv = 512 * 96 + 1024 * 512;
    prep_weights<<<(nconv + 255) / 256, 256, 0, stream>>>(W1, W2, W1h, W2h);

    const int S = in_sizes[1] / 2;              // 1024 segments
    gravity_fused<<<S, 256, LDS_BYTES, stream>>>(h_state, sse, cbr, bm, om, Wsm, bsv,
                                                 g1, be1, g2, be2, W1h, W2h, out);
}
